// EncoderLayer_68642167324917
// MI455X (gfx1250) — matrix operand order repaired and checked by reading, not compiled, not run on hardware
//
#include <hip/hip_runtime.h>

#define DEVINL __device__ __forceinline__

typedef __bf16 bf16_t;
typedef __attribute__((ext_vector_type(16))) __bf16 v16bf;
typedef __attribute__((ext_vector_type(8)))  __bf16 v8bf;
typedef __attribute__((ext_vector_type(8)))  float  v8f;

constexpr int B_  = 4;
constexpr int S_  = 2048;
constexpr int D_  = 1024;
constexpr int H_  = 16;
constexpr int FF_ = 4096;
constexpr int DKH = 64;                 // head dim
constexpr int MROWS = B_ * S_;          // 8192 token rows

// ---------------------------------------------------------------------------
// CDNA5 async copy: global -> LDS, 16B per lane, tracked by ASYNCcnt.
// VDST = LDS byte address (low 32 bits of generic shared pointer; the flat
// LDS aperture truncates to addr[31:0] and HW adds LDS_BASE).
// ---------------------------------------------------------------------------
DEVINL void async_load_b128(void* lds_ptr, const void* gptr) {
    const unsigned int lds_off = (unsigned int)(size_t)lds_ptr;
    asm volatile("global_load_async_to_lds_b128 %0, %1, off"
                 :: "v"(lds_off), "v"(gptr) : "memory");
}
DEVINL void wait_async0() {
    asm volatile("s_wait_asynccnt 0" ::: "memory");
}

// ---------------------------------------------------------------------------
// WMMA fragment helpers (bf16 16x16x32, wave32)
//  A operand: lane(0..15)=row M, lane-half selects K chunks {lh*8..lh*8+7} and
//             {16+lh*8..}; packed 2 bf16 per VGPR (ISA 7.12.2).
//  B operand: lane(0..15)=col N, lane-half selects K chunks {lh*8..lh*8+7} and
//             {16+lh*8..} (same as A).
// ---------------------------------------------------------------------------
DEVINL v16bf load_a_frag(const bf16_t* rp, int lh) {
    union { v16bf v; v8bf h[2]; } u;
    u.h[0] = *(const v8bf*)(rp + lh * 8);
    u.h[1] = *(const v8bf*)(rp + 16 + lh * 8);
    return u.v;
}
DEVINL v16bf load_b_frag(const bf16_t* rp, int lh) {
    union { v16bf v; v8bf h[2]; } u;
    u.h[0] = *(const v8bf*)(rp + lh * 8);
    u.h[1] = *(const v8bf*)(rp + 16 + lh * 8);
    return u.v;
}
DEVINL v8f wmma_bf16(v16bf a, v16bf b, v8f c) {
    return __builtin_amdgcn_wmma_f32_16x16x32_bf16(
        false, a, false, b, (short)0, c, false, false);
}

// ---------------------------------------------------------------------------
// Generic GEMM: C[M,N] = A[M,K] (bf16) x Bt[N,K]^T (bf16) + bias
//  optional ReLU, optional fp32 residual add; fp32 and/or bf16 output.
//  Block tile 128x128, 8 waves; wave tile 64x32 (4x2 WMMA tiles), K step 32.
//  Double-buffered LDS fed by async global->LDS copies.
// ---------------------------------------------------------------------------
constexpr int TM = 128, TN = 128, TK = 32, GP = 48;   // GP*2B = 96B row pitch

template <bool RELU, bool RESID>
__global__ __launch_bounds__(256, 1)
void gemm_kernel(const bf16_t* __restrict__ A, const bf16_t* __restrict__ Bt,
                 const float* __restrict__ bias, const float* __restrict__ resid,
                 float* __restrict__ outF, bf16_t* __restrict__ outB,
                 int M, int N, int K)
{
    __shared__ bf16_t As[2][TM * GP];
    __shared__ bf16_t Bs[2][TN * GP];

    const int tid  = threadIdx.x;
    const int lane = tid & 31, wid = tid >> 5;
    const int lm   = lane & 15, lh = lane >> 4;
    const int bm   = blockIdx.y * TM;
    const int bn   = blockIdx.x * TN;
    const int wm   = (wid & 1) * 64;
    const int wn   = (wid >> 1) * 32;

    v8f acc[4][2] = {};

    const int r0 = tid >> 2;            // 0..63
    const int kk = (tid & 3) * 8;       // 0,8,16,24

    // issue one 128x32 A-tile + B-tile into LDS buffer `buf` (async)
    auto issue_tile = [&](int buf, int k0) {
        async_load_b128(&As[buf][(r0     ) * GP + kk],
                        A  + (size_t)(bm + r0     ) * K + k0 + kk);
        async_load_b128(&As[buf][(r0 + 64) * GP + kk],
                        A  + (size_t)(bm + r0 + 64) * K + k0 + kk);
        async_load_b128(&Bs[buf][(r0     ) * GP + kk],
                        Bt + (size_t)(bn + r0     ) * K + k0 + kk);
        async_load_b128(&Bs[buf][(r0 + 64) * GP + kk],
                        Bt + (size_t)(bn + r0 + 64) * K + k0 + kk);
    };

    issue_tile(0, 0);
    wait_async0();
    __syncthreads();

    int buf = 0;
    for (int k0 = 0; k0 < K; k0 += TK, buf ^= 1) {
        const bool more = (k0 + TK) < K;
        if (more) issue_tile(buf ^ 1, k0 + TK);   // prefetch next tile

        v16bf af[4], bfr[2];
#pragma unroll
        for (int t = 0; t < 4; ++t)
            af[t] = load_a_frag(&As[buf][(wm + t * 16 + lm) * GP], lh);
#pragma unroll
        for (int j = 0; j < 2; ++j)
            bfr[j] = load_b_frag(&Bs[buf][(wn + j * 16 + lm) * GP], lh);

#pragma unroll
        for (int t = 0; t < 4; ++t)
#pragma unroll
            for (int j = 0; j < 2; ++j)
                acc[t][j] = wmma_bf16(af[t], bfr[j], acc[t][j]);

        if (more) wait_async0();                  // next tile landed (this wave)
        __syncthreads();                          // ...and all other waves
    }

    // epilogue: C layout is row = v + 8*lh, col = lm within each 16x16 tile
#pragma unroll
    for (int t = 0; t < 4; ++t) {
#pragma unroll
        for (int j = 0; j < 2; ++j) {
            const int col = bn + wn + j * 16 + lm;
            const float bv = bias[col];
#pragma unroll
            for (int v = 0; v < 8; ++v) {
                const int row = bm + wm + t * 16 + v + 8 * lh;
                float val = acc[t][j][v] + bv;
                if (RELU)  val = fmaxf(val, 0.0f);
                if (RESID) val += resid[(size_t)row * N + col];
                if (outF)  outF[(size_t)row * N + col] = val;
                if (outB)  outB[(size_t)row * N + col] = (bf16_t)val;
            }
        }
    }
}

// ---------------------------------------------------------------------------
// Flash attention: one workgroup = (b,h) x 64-query tile, 4 waves.
// Each wave owns 16 query rows. Streams 64-key tiles with online softmax.
// Q/K tiles arrive via async global->LDS; V is loaded through VGPRs because
// it must be transposed into LDS for the P@V B-operand.
// ---------------------------------------------------------------------------
constexpr int AP = 72;                  // LDS pitch elems (144B, 16B aligned)

__global__ __launch_bounds__(128, 1)
void attn_kernel(const bf16_t* __restrict__ Q, const bf16_t* __restrict__ Km,
                 const bf16_t* __restrict__ V, const int* __restrict__ mask,
                 bf16_t* __restrict__ O)
{
    __shared__ bf16_t Qs[64 * AP];
    __shared__ bf16_t Ks[64 * AP];
    __shared__ bf16_t Vt[64 * AP];      // transposed: [dk][key]
    __shared__ bf16_t Ps[64 * AP];      // probabilities (per-wave rows)

    const int tid  = threadIdx.x;
    const int lane = tid & 31, wid = tid >> 5;
    const int lm   = lane & 15, lh = lane >> 4;
    const int bh = blockIdx.y;
    const int b  = bh / H_, h = bh % H_;
    const int q0 = blockIdx.x * 64;
    const size_t headoff = (size_t)b * S_ * D_ + (size_t)h * DKH;

    // load 64x64 Q tile (async)
#pragma unroll
    for (int c = 0; c < 4; ++c) {
        const int ch = tid + c * 128;
        const int r = ch >> 3, kk = (ch & 7) * 8;
        async_load_b128(&Qs[r * AP + kk],
                        Q + headoff + (size_t)(q0 + r) * D_ + kk);
    }
    wait_async0();
    __syncthreads();

    v16bf aq0 = load_a_frag(&Qs[(wid * 16 + lm) * AP], lh);        // k 0..31
    v16bf aq1 = load_a_frag(&Qs[(wid * 16 + lm) * AP + 32], lh);   // k 32..63

    float m_i[8], l_i[8];
#pragma unroll
    for (int v = 0; v < 8; ++v) { m_i[v] = -1e30f; l_i[v] = 0.0f; }
    v8f acc_o[4] = {};

    for (int kt = 0; kt < S_; kt += 64) {
        __syncthreads();                 // prior-iter Ks/Vt consumers done
#pragma unroll
        for (int c = 0; c < 4; ++c) {
            const int ch = tid + c * 128;
            const int r = ch >> 3, kk = (ch & 7) * 8;
            async_load_b128(&Ks[r * AP + kk],
                            Km + headoff + (size_t)(kt + r) * D_ + kk);
            v8bf vv = *(const v8bf*)(V + headoff + (size_t)(kt + r) * D_ + kk);
#pragma unroll
            for (int i = 0; i < 8; ++i)
                Vt[(kk + i) * AP + r] = vv[i];                      // transpose
        }
        wait_async0();
        __syncthreads();

        // S = Q K^T (scaled later)
        v8f s[4];
#pragma unroll
        for (int j = 0; j < 4; ++j) {
            v16bf bk0 = load_b_frag(&Ks[(j * 16 + lm) * AP], lh);
            v16bf bk1 = load_b_frag(&Ks[(j * 16 + lm) * AP + 32], lh);
            v8f z = {};
            z = wmma_bf16(aq0, bk0, z);
            s[j] = wmma_bf16(aq1, bk1, z);
        }

        int msk[4];
#pragma unroll
        for (int j = 0; j < 4; ++j)
            msk[j] = mask[(size_t)b * S_ + kt + j * 16 + lm];

        // online softmax per row (row = v + 8*lh of this wave's 16 q rows)
#pragma unroll
        for (int v = 0; v < 8; ++v) {
            float mx = -1e30f;
#pragma unroll
            for (int j = 0; j < 4; ++j) {
                float sv = s[j][v] * 0.125f;           // 1/sqrt(64)
                sv = msk[j] ? sv : -1e9f;
                s[j][v] = sv;
                mx = fmaxf(mx, sv);
            }
            mx = fmaxf(mx, __shfl_xor(mx, 1));
            mx = fmaxf(mx, __shfl_xor(mx, 2));
            mx = fmaxf(mx, __shfl_xor(mx, 4));
            mx = fmaxf(mx, __shfl_xor(mx, 8));
            const float mnew = fmaxf(m_i[v], mx);
            const float corr = __expf(m_i[v] - mnew);
            float rs = 0.0f;
#pragma unroll
            for (int j = 0; j < 4; ++j) {
                const float p = __expf(s[j][v] - mnew);
                rs += p;
                Ps[(wid * 16 + v + 8 * lh) * AP + (j * 16 + lm)] = (bf16_t)p;
            }
            rs += __shfl_xor(rs, 1);
            rs += __shfl_xor(rs, 2);
            rs += __shfl_xor(rs, 4);
            rs += __shfl_xor(rs, 8);
            l_i[v] = l_i[v] * corr + rs;
            m_i[v] = mnew;
#pragma unroll
            for (int t = 0; t < 4; ++t)
                acc_o[t][v] *= corr;
        }

        // O += P V  (Ps rows are wave-private; same-wave LDS is in-order)
        v16bf ap0 = load_a_frag(&Ps[(wid * 16 + lm) * AP], lh);
        v16bf ap1 = load_a_frag(&Ps[(wid * 16 + lm) * AP + 32], lh);
#pragma unroll
        for (int t = 0; t < 4; ++t) {
            v16bf bv0 = load_b_frag(&Vt[(t * 16 + lm) * AP], lh);
            v16bf bv1 = load_b_frag(&Vt[(t * 16 + lm) * AP + 32], lh);
            acc_o[t] = wmma_bf16(ap0, bv0, acc_o[t]);
            acc_o[t] = wmma_bf16(ap1, bv1, acc_o[t]);
        }
    }

#pragma unroll
    for (int t = 0; t < 4; ++t)
#pragma unroll
        for (int v = 0; v < 8; ++v) {
            const int row = q0 + wid * 16 + v + 8 * lh;
            O[((size_t)b * S_ + row) * D_ + h * DKH + t * 16 + lm] =
                (bf16_t)(acc_o[t][v] / l_i[v]);
        }
}

// ---------------------------------------------------------------------------
// LayerNorm (faithful: ddof=1, eps added to std, scalar alpha/beta)
// one 256-thread block per row of 1024
// ---------------------------------------------------------------------------
__global__ __launch_bounds__(256, 1)
void layernorm_kernel(const float* __restrict__ in, const float* __restrict__ alpha,
                      const float* __restrict__ beta, float* __restrict__ outF,
                      bf16_t* __restrict__ outB)
{
    __shared__ float red[256];
    const int tid = threadIdx.x;
    const size_t row = blockIdx.x;
    const float* x = in + row * D_;

    float v[4];
    float s = 0.0f;
#pragma unroll
    for (int i = 0; i < 4; ++i) { v[i] = x[tid + i * 256]; s += v[i]; }
    red[tid] = s; __syncthreads();
    for (int o = 128; o > 0; o >>= 1) { if (tid < o) red[tid] += red[tid + o]; __syncthreads(); }
    const float mean = red[0] / (float)D_;
    __syncthreads();

    float sq = 0.0f;
#pragma unroll
    for (int i = 0; i < 4; ++i) { const float d = v[i] - mean; sq += d * d; }
    red[tid] = sq; __syncthreads();
    for (int o = 128; o > 0; o >>= 1) { if (tid < o) red[tid] += red[tid + o]; __syncthreads(); }
    const float var = red[0] / (float)(D_ - 1);
    const float inv = alpha[0] / (sqrtf(var) + 1e-6f);
    const float bb  = beta[0];

#pragma unroll
    for (int i = 0; i < 4; ++i) {
        const float o = (v[i] - mean) * inv + bb;
        const int c = tid + i * 256;
        if (outF) outF[row * D_ + c] = o;
        if (outB) outB[row * D_ + c] = (bf16_t)o;
    }
}

// ---------------------------------------------------------------------------
// fp32 W[K,N] -> bf16 Wt[N,K] via 32x32 LDS tile (coalesced both directions)
// ---------------------------------------------------------------------------
__global__ __launch_bounds__(256, 1)
void transpose_cast_kernel(const float* __restrict__ W, bf16_t* __restrict__ Wt,
                           int K, int N)
{
    __shared__ float t[32][33];
    const int tx = threadIdx.x & 31, ty = threadIdx.x >> 5;  // 32x8
    const int k0 = blockIdx.y * 32, n0 = blockIdx.x * 32;
#pragma unroll
    for (int i = 0; i < 4; ++i)
        t[ty + i * 8][tx] = W[(size_t)(k0 + ty + i * 8) * N + n0 + tx];
    __syncthreads();
#pragma unroll
    for (int i = 0; i < 4; ++i)
        Wt[(size_t)(n0 + ty + i * 8) * K + k0 + tx] = (bf16_t)t[tx][ty + i * 8];
}

__global__ __launch_bounds__(256, 1)
void cast_kernel(const float* __restrict__ in, bf16_t* __restrict__ out)
{
    const size_t i = (size_t)blockIdx.x * 256 + threadIdx.x;
    out[i] = (bf16_t)in[i];
}

// ---------------------------------------------------------------------------
extern "C" void kernel_launch(void* const* d_in, const int* in_sizes, int n_in,
                              void* d_out, int out_size, void* d_ws, size_t ws_size,
                              hipStream_t stream)
{
    (void)in_sizes; (void)n_in; (void)out_size; (void)ws_size;

    const float* x      = (const float*)d_in[0];
    const int*   mask   = (const int*)  d_in[1];
    const float* Wq     = (const float*)d_in[2];
    const float* bq     = (const float*)d_in[3];
    const float* Wk     = (const float*)d_in[4];
    const float* bk     = (const float*)d_in[5];
    const float* Wv     = (const float*)d_in[6];
    const float* bv     = (const float*)d_in[7];
    const float* Wo     = (const float*)d_in[8];
    const float* bo     = (const float*)d_in[9];
    const float* W1     = (const float*)d_in[10];
    const float* b1     = (const float*)d_in[11];
    const float* W2     = (const float*)d_in[12];
    const float* b2     = (const float*)d_in[13];
    const float* alpha1 = (const float*)d_in[14];
    const float* beta1  = (const float*)d_in[15];
    const float* alpha2 = (const float*)d_in[16];
    const float* beta2  = (const float*)d_in[17];
    float* out = (float*)d_out;

    char* w = (char*)d_ws;
    size_t off = 0;
    auto take = [&](size_t bytes) {
        char* p = w + off;
        off += (bytes + 255) & ~(size_t)255;
        return p;
    };

    bf16_t* xb    = (bf16_t*)take((size_t)MROWS * D_ * 2);
    bf16_t* WqT   = (bf16_t*)take((size_t)D_ * D_ * 2);
    bf16_t* WkT   = (bf16_t*)take((size_t)D_ * D_ * 2);
    bf16_t* WvT   = (bf16_t*)take((size_t)D_ * D_ * 2);
    bf16_t* WoT   = (bf16_t*)take((size_t)D_ * D_ * 2);
    bf16_t* W1T   = (bf16_t*)take((size_t)D_ * FF_ * 2);
    bf16_t* W2T   = (bf16_t*)take((size_t)FF_ * D_ * 2);
    bf16_t* Qb    = (bf16_t*)take((size_t)MROWS * D_ * 2);
    bf16_t* Kb    = (bf16_t*)take((size_t)MROWS * D_ * 2);
    bf16_t* Vb    = (bf16_t*)take((size_t)MROWS * D_ * 2);
    bf16_t* attnb = (bf16_t*)take((size_t)MROWS * D_ * 2);
    float*  y1    = (float*) take((size_t)MROWS * D_ * 4);
    float*  ln1f  = (float*) take((size_t)MROWS * D_ * 4);
    bf16_t* ln1b  = (bf16_t*)take((size_t)MROWS * D_ * 2);
    // aliased buffers (lifetimes disjoint): h1 reuses Qb..attnb, y2 reuses y1
    bf16_t* h1 = Qb;        // needs 8192*4096*2 = 64MB == Qb+Kb+Vb+attnb span
    float*  y2 = y1;

    // --- weight prep + input cast ---
    transpose_cast_kernel<<<dim3(D_ / 32,  D_ / 32),  256, 0, stream>>>(Wq, WqT, D_, D_);
    transpose_cast_kernel<<<dim3(D_ / 32,  D_ / 32),  256, 0, stream>>>(Wk, WkT, D_, D_);
    transpose_cast_kernel<<<dim3(D_ / 32,  D_ / 32),  256, 0, stream>>>(Wv, WvT, D_, D_);
    transpose_cast_kernel<<<dim3(D_ / 32,  D_ / 32),  256, 0, stream>>>(Wo, WoT, D_, D_);
    transpose_cast_kernel<<<dim3(FF_ / 32, D_ / 32),  256, 0, stream>>>(W1, W1T, D_, FF_);
    transpose_cast_kernel<<<dim3(D_ / 32,  FF_ / 32), 256, 0, stream>>>(W2, W2T, FF_, D_);
    cast_kernel<<<(MROWS * (size_t)D_) / 256, 256, 0, stream>>>(x, xb);

    // --- QKV projections ---
    gemm_kernel<false, false><<<dim3(D_ / 128, MROWS / 128), 256, 0, stream>>>(
        xb, WqT, bq, nullptr, nullptr, Qb, MROWS, D_, D_);
    gemm_kernel<false, false><<<dim3(D_ / 128, MROWS / 128), 256, 0, stream>>>(
        xb, WkT, bk, nullptr, nullptr, Kb, MROWS, D_, D_);
    gemm_kernel<false, false><<<dim3(D_ / 128, MROWS / 128), 256, 0, stream>>>(
        xb, WvT, bv, nullptr, nullptr, Vb, MROWS, D_, D_);

    // --- attention ---
    attn_kernel<<<dim3(S_ / 64, B_ * H_), 128, 0, stream>>>(Qb, Kb, Vb, mask, attnb);

    // --- output projection + residual, LN1 ---
    gemm_kernel<false, true><<<dim3(D_ / 128, MROWS / 128), 256, 0, stream>>>(
        attnb, WoT, bo, x, y1, nullptr, MROWS, D_, D_);
    layernorm_kernel<<<MROWS, 256, 0, stream>>>(y1, alpha1, beta1, ln1f, ln1b);

    // --- FFN + residual, LN2 ---
    gemm_kernel<true, false><<<dim3(FF_ / 128, MROWS / 128), 256, 0, stream>>>(
        ln1b, W1T, b1, nullptr, nullptr, h1, MROWS, FF_, D_);
    gemm_kernel<false, true><<<dim3(D_ / 128, MROWS / 128), 256, 0, stream>>>(
        h1, W2T, b2, ln1f, y2, nullptr, MROWS, D_, FF_);
    layernorm_kernel<<<MROWS, 256, 0, stream>>>(y2, alpha2, beta2, out, nullptr);
}
